// SR_GNN_att_agg_42253888258364
// MI455X (gfx1250) — matrix operand order repaired and checked by reading, not compiled, not run on hardware
//
#include <hip/hip_runtime.h>
#include <math.h>

typedef __attribute__((ext_vector_type(16))) __bf16 v16bf;
typedef __attribute__((ext_vector_type(8)))  float  v8f;

#define N_NODES   65536
#define N_EDGES   131072
#define N_GRAPHS  4096
#define HDIM      128
#define DDIM      64
#define NUM_ITEMS 50000
#define IN_F      320   // feature dim without the price column

// ---------- helpers ----------

__device__ __forceinline__ unsigned short f2bf(float f) {
  unsigned int x = __float_as_uint(f);
  unsigned int r = x + 0x7fffu + ((x >> 16) & 1u);   // round-to-nearest-even
  return (unsigned short)(r >> 16);
}

// A fragment (16x32 bf16), row-major source in LDS, per ISA 16-bit A layout:
// lanes 0-15: V0-3 = K 0..7 pairs, V4-7 = K 16..23; lanes 16-31: +8 / +24.
__device__ __forceinline__ v16bf load_a_lds(const unsigned short* row, int kt, int half) {
  union { unsigned int u[8]; v16bf v; } a;
  int b1 = kt + half * 8;
  int b2 = kt + 16 + half * 8;
#pragma unroll
  for (int i = 0; i < 4; ++i) {
    a.u[i]     = (unsigned)row[b1 + 2 * i] | ((unsigned)row[b1 + 2 * i + 1] << 16);
    a.u[4 + i] = (unsigned)row[b2 + 2 * i] | ((unsigned)row[b2 + 2 * i + 1] << 16);
  }
  return a.v;
}

// B fragment (32x16 bf16). W is (out x K) row-major bf16; B[k][n]=W[col][k].
// lanes 0-15 hold K=kt..kt+7 and kt+16..kt+23 for col=lane, lanes 16-31 hold +8 / +24.
// => each lane reads two 8-element bf16 chunks from its output row.
__device__ __forceinline__ v16bf load_b_row(const unsigned short* W, int K, int col,
                                            int kt, int half) {
  const unsigned short* p = W + (size_t)col * K + kt + half * 8;
  union { uint4 q[2]; v16bf v; } b;
  b.q[0] = *(const uint4*)(p);
  b.q[1] = *(const uint4*)(p + 16);
  return b.v;
}

__device__ __forceinline__ float sigmoidf_(float x) { return 1.0f / (1.0f + __expf(-x)); }

// ---------- prep kernels ----------

__global__ void k_convert(const float* __restrict__ src, unsigned short* __restrict__ dst, int n) {
  for (int i = blockIdx.x * blockDim.x + threadIdx.x; i < n; i += gridDim.x * blockDim.x)
    dst[i] = f2bf(src[i]);
}

// Wm is (128 x 321); drop column 0 (price) -> (128 x 320) bf16
__global__ void k_convert_wm(const float* __restrict__ Wm, unsigned short* __restrict__ dst) {
  int n = HDIM * IN_F;
  for (int i = blockIdx.x * blockDim.x + threadIdx.x; i < n; i += gridDim.x * blockDim.x) {
    int o = i / IN_F, k = i - o * IN_F;
    dst[i] = f2bf(Wm[(size_t)o * 321 + 1 + k]);
  }
}

__global__ void k_zero(float* __restrict__ p, int n) {
  for (int i = blockIdx.x * blockDim.x + threadIdx.x; i < n; i += gridDim.x * blockDim.x)
    p[i] = 0.0f;
}

// ---------- stage 1: embedding gather + message linear (WMMA) ----------
// h[n] = [price, cat, sub, el, br, item] @ Wm^T + bm
__global__ __launch_bounds__(256) void k_message(
    const int* __restrict__ cat, const int* __restrict__ sub, const int* __restrict__ el,
    const int* __restrict__ br, const int* __restrict__ pid, const float* __restrict__ price,
    const float* __restrict__ ecat, const float* __restrict__ esub, const float* __restrict__ eel,
    const float* __restrict__ ebr, const float* __restrict__ eitem,
    const unsigned short* __restrict__ Wm_bf, const float* __restrict__ Wm,
    const float* __restrict__ bm, float* __restrict__ h) {
  __shared__ unsigned short xb[16 * IN_F];
  __shared__ float pr[16];
  int tid = threadIdx.x;
  int node0 = blockIdx.x * 16;
  if (tid < 16) pr[tid] = price[node0 + tid];
  for (int idx = tid; idx < 16 * IN_F; idx += 256) {
    int n = idx / IN_F, c = idx - n * IN_F;
    int node = node0 + n;
    int t = c >> 6, o = c & 63;
    float v;
    if      (t == 0) v = ecat[(size_t)cat[node] * DDIM + o];
    else if (t == 1) v = esub[(size_t)sub[node] * DDIM + o];
    else if (t == 2) v = eel [(size_t)el [node] * DDIM + o];
    else if (t == 3) v = ebr [(size_t)br [node] * DDIM + o];
    else             v = eitem[(size_t)pid[node] * DDIM + o];
    xb[idx] = f2bf(v);
  }
  __syncthreads();
  int w = tid >> 5, lane = tid & 31;
  int half = lane >> 4, l16 = lane & 15;
  int col = w * 16 + l16;                       // output channel (H=128 = 8 tiles)
  const unsigned short* arow = xb + l16 * IN_F;
  v8f acc = {};
#pragma unroll
  for (int kt = 0; kt < IN_F; kt += 32) {
    v16bf a = load_a_lds(arow, kt, half);
    v16bf b = load_b_row(Wm_bf, IN_F, col, kt, half);
    acc = __builtin_amdgcn_wmma_f32_16x16x32_bf16(false, a, false, b, (short)0, acc, false, false);
  }
  float bias = bm[col];
  float w0 = Wm[(size_t)col * 321];             // price column
#pragma unroll
  for (int r = 0; r < 8; ++r) {
    int mrow = r + 8 * half;
    h[(size_t)(node0 + mrow) * HDIM + col] = acc[r] + bias + pr[mrow] * w0;
  }
}

// ---------- stage 2: scatter-mean over edges ----------

__global__ void k_edge_count(const int* __restrict__ ei, float* __restrict__ cnt) {
  int e = blockIdx.x * blockDim.x + threadIdx.x;
  if (e < N_EDGES) atomicAdd(&cnt[ei[N_EDGES + e]], 1.0f);
}

__global__ void k_edge_scatter(const int* __restrict__ ei, const float* __restrict__ h,
                               float* __restrict__ m) {
  int g = blockIdx.x * blockDim.x + threadIdx.x;
  int e = g >> 7, c = g & 127;
  if (e < N_EDGES)
    atomicAdd(&m[(size_t)ei[N_EDGES + e] * HDIM + c], h[(size_t)ei[e] * HDIM + c]);
}

__global__ void k_normalize(float* __restrict__ m, const float* __restrict__ cnt) {
  int i = blockIdx.x * blockDim.x + threadIdx.x;
  if (i < N_NODES * HDIM) m[i] *= 1.0f / fmaxf(cnt[i >> 7], 1.0f);
}

// ---------- stage 3: fused GRU cell (two 128->384 GEMMs in LDS, WMMA) ----------

__global__ __launch_bounds__(256) void k_gru(
    const float* __restrict__ mglob, float* __restrict__ hglob,
    const unsigned short* __restrict__ Wih_bf, const unsigned short* __restrict__ Whh_bf,
    const float* __restrict__ b_ih, const float* __restrict__ b_hh) {
  __shared__ unsigned short mb[16 * HDIM];
  __shared__ unsigned short hb[16 * HDIM];
  __shared__ float gi[16 * 384];
  __shared__ float gh[16 * 384];
  int tid = threadIdx.x;
  int node0 = blockIdx.x * 16;
  for (int idx = tid; idx < 16 * HDIM; idx += 256) {
    int n = idx >> 7, c = idx & 127;
    size_t gidx = (size_t)(node0 + n) * HDIM + c;
    mb[idx] = f2bf(mglob[gidx]);
    hb[idx] = f2bf(hglob[gidx]);
  }
  __syncthreads();
  int w = tid >> 5, lane = tid & 31;
  int half = lane >> 4, l16 = lane & 15;
#pragma unroll
  for (int i = 0; i < 6; ++i) {                  // 48 tiles = 2 GEMMs x 24 tiles, 8 waves
    int t = w + 8 * i;                           // uniform per wave
    const unsigned short* A = (t < 24) ? mb : hb;
    const unsigned short* B = (t < 24) ? Wih_bf : Whh_bf;
    float* out = (t < 24) ? gi : gh;
    int cb = ((t < 24) ? t : (t - 24)) * 16;
    int col = cb + l16;
    const unsigned short* arow = A + l16 * HDIM;
    v8f acc = {};
#pragma unroll
    for (int kt = 0; kt < HDIM; kt += 32) {
      v16bf a = load_a_lds(arow, kt, half);
      v16bf b = load_b_row(B, HDIM, col, kt, half);
      acc = __builtin_amdgcn_wmma_f32_16x16x32_bf16(false, a, false, b, (short)0, acc, false, false);
    }
#pragma unroll
    for (int r = 0; r < 8; ++r) out[(r + 8 * half) * 384 + col] = acc[r];
  }
  __syncthreads();
  for (int idx = tid; idx < 16 * HDIM; idx += 256) {
    int n = idx >> 7, c = idx & 127;
    const float* gip = gi + n * 384;
    const float* ghp = gh + n * 384;
    float r  = sigmoidf_(gip[c] + b_ih[c] + ghp[c] + b_hh[c]);
    float z  = sigmoidf_(gip[HDIM + c] + b_ih[HDIM + c] + ghp[HDIM + c] + b_hh[HDIM + c]);
    float nn = tanhf(gip[2 * HDIM + c] + b_ih[2 * HDIM + c] +
                     r * (ghp[2 * HDIM + c] + b_hh[2 * HDIM + c]));
    size_t gidx = (size_t)(node0 + n) * HDIM + c;
    float hold = hglob[gidx];
    hglob[gidx] = (1.0f - z) * nn + z * hold;
  }
}

// ---------- stage 4: attention gate (WMMA) ----------

__global__ __launch_bounds__(256) void k_attn(
    const float* __restrict__ hglob, const unsigned short* __restrict__ Wg1_bf,
    const float* __restrict__ bg1, const float* __restrict__ Wg2,
    const float* __restrict__ bg2, float* __restrict__ gate) {
  __shared__ unsigned short hb[16 * HDIM];
  __shared__ float tt[16 * HDIM];
  int tid = threadIdx.x;
  int node0 = blockIdx.x * 16;
  for (int idx = tid; idx < 16 * HDIM; idx += 256) {
    int n = idx >> 7, c = idx & 127;
    hb[idx] = f2bf(hglob[(size_t)(node0 + n) * HDIM + c]);
  }
  __syncthreads();
  int w = tid >> 5, lane = tid & 31;
  int half = lane >> 4, l16 = lane & 15;
  int col = w * 16 + l16;
  const unsigned short* arow = hb + l16 * HDIM;
  v8f acc = {};
#pragma unroll
  for (int kt = 0; kt < HDIM; kt += 32) {
    v16bf a = load_a_lds(arow, kt, half);
    v16bf b = load_b_row(Wg1_bf, HDIM, col, kt, half);
    acc = __builtin_amdgcn_wmma_f32_16x16x32_bf16(false, a, false, b, (short)0, acc, false, false);
  }
  float bias = bg1[col];
#pragma unroll
  for (int r = 0; r < 8; ++r)
    tt[(r + 8 * half) * HDIM + col] = fmaxf(acc[r] + bias, 0.0f);
  __syncthreads();
  if (tid < 16) {
    float s = 0.0f;
    for (int c = 0; c < HDIM; ++c) s += tt[tid * HDIM + c] * Wg2[c];
    gate[node0 + tid] = s + bg2[0];
  }
}

// ---------- stage 5: segment softmax + weighted pool (16 nodes/graph, contiguous) ----------

__global__ __launch_bounds__(128) void k_softpool(const float* __restrict__ gate,
                                                  const float* __restrict__ h,
                                                  float* __restrict__ pooled) {
  __shared__ float sg[16];
  __shared__ float sa[16];
  int g = blockIdx.x, tid = threadIdx.x;
  if (tid < 16) sg[tid] = gate[g * 16 + tid];
  __syncthreads();
  float mx = sg[0];
  for (int i = 1; i < 16; ++i) mx = fmaxf(mx, sg[i]);
  if (tid < 16) sa[tid] = __expf(sg[tid] - mx);
  __syncthreads();
  float den = 0.0f;
  for (int i = 0; i < 16; ++i) den += sa[i];
  float inv = 1.0f / den;
  float acc = 0.0f;
  for (int n = 0; n < 16; ++n)
    acc += sa[n] * inv * h[(size_t)(g * 16 + n) * HDIM + tid];
  pooled[(size_t)g * HDIM + tid] = acc;
}

// ---------- stage 6: final FC pooled(4096x128) @ Wfc^T(128x50000) + bfc (WMMA) ----------

__global__ __launch_bounds__(256) void k_fc(const float* __restrict__ pooled,
                                            const unsigned short* __restrict__ Wfc_bf,
                                            const float* __restrict__ bfc,
                                            float* __restrict__ out) {
  __shared__ unsigned short pb[16 * HDIM];
  int tid = threadIdx.x;
  int gb = blockIdx.x;                           // graph tile (16 rows)
  for (int idx = tid; idx < 16 * HDIM; idx += 256)
    pb[idx] = f2bf(pooled[(size_t)gb * 16 * HDIM + idx]);
  __syncthreads();
  int w = tid >> 5, lane = tid & 31;
  int half = lane >> 4, l16 = lane & 15;
  int cb = (blockIdx.y * 8 + w) * 16;            // uniform per wave
  if (cb >= NUM_ITEMS) return;
  int col = cb + l16;
  const unsigned short* arow = pb + l16 * HDIM;
  v8f acc = {};
#pragma unroll
  for (int kt = 0; kt < HDIM; kt += 32) {
    v16bf a = load_a_lds(arow, kt, half);
    v16bf b = load_b_row(Wfc_bf, HDIM, col, kt, half);
    acc = __builtin_amdgcn_wmma_f32_16x16x32_bf16(false, a, false, b, (short)0, acc, false, false);
  }
  float bias = bfc[col];
#pragma unroll
  for (int r = 0; r < 8; ++r)
    out[(size_t)(gb * 16 + r + 8 * half) * NUM_ITEMS + col] = acc[r] + bias;
}

// ---------- launcher ----------

extern "C" void kernel_launch(void* const* d_in, const int* in_sizes, int n_in,
                              void* d_out, int out_size, void* d_ws, size_t ws_size,
                              hipStream_t stream) {
  (void)in_sizes; (void)n_in; (void)out_size; (void)ws_size;
  const int*   cat   = (const int*)d_in[0];
  const int*   sub   = (const int*)d_in[1];
  const int*   el    = (const int*)d_in[2];
  const int*   br    = (const int*)d_in[3];
  const int*   pid   = (const int*)d_in[4];
  const float* price = (const float*)d_in[5];
  const int*   ei    = (const int*)d_in[6];
  /* d_in[7] = batch: unused, graphs are contiguous 16-node blocks */
  const float* ecat  = (const float*)d_in[8];
  const float* esub  = (const float*)d_in[9];
  const float* eel   = (const float*)d_in[10];
  const float* ebr   = (const float*)d_in[11];
  const float* eitem = (const float*)d_in[12];
  const float* Wm    = (const float*)d_in[13];
  const float* bm    = (const float*)d_in[14];
  const float* W_ih  = (const float*)d_in[15];
  const float* b_ih  = (const float*)d_in[16];
  const float* W_hh  = (const float*)d_in[17];
  const float* b_hh  = (const float*)d_in[18];
  const float* Wg1   = (const float*)d_in[19];
  const float* bg1   = (const float*)d_in[20];
  const float* Wg2   = (const float*)d_in[21];
  const float* bg2   = (const float*)d_in[22];
  const float* Wfc   = (const float*)d_in[23];
  const float* bfc   = (const float*)d_in[24];
  float* out = (float*)d_out;

  char* ws = (char*)d_ws;
  size_t off = 0;
  float* h      = (float*)(ws + off); off += (size_t)N_NODES * HDIM * 4;
  float* m      = (float*)(ws + off); off += (size_t)N_NODES * HDIM * 4;
  float* cnt    = (float*)(ws + off); off += (size_t)N_NODES * 4;
  float* gate   = (float*)(ws + off); off += (size_t)N_NODES * 4;
  float* pooled = (float*)(ws + off); off += (size_t)N_GRAPHS * HDIM * 4;
  unsigned short* Wm_bf  = (unsigned short*)(ws + off); off += (size_t)HDIM * IN_F * 2;
  unsigned short* Wih_bf = (unsigned short*)(ws + off); off += (size_t)3 * HDIM * HDIM * 2;
  unsigned short* Whh_bf = (unsigned short*)(ws + off); off += (size_t)3 * HDIM * HDIM * 2;
  unsigned short* Wg1_bf = (unsigned short*)(ws + off); off += (size_t)HDIM * HDIM * 2;
  unsigned short* Wfc_bf = (unsigned short*)(ws + off); off += (size_t)NUM_ITEMS * HDIM * 2;

  // Weight bf16 conversion (re-done every call; deterministic)
  k_convert_wm<<<160, 256, 0, stream>>>(Wm, Wm_bf);
  k_convert<<<192, 256, 0, stream>>>(W_ih, Wih_bf, 3 * HDIM * HDIM);
  k_convert<<<192, 256, 0, stream>>>(W_hh, Whh_bf, 3 * HDIM * HDIM);
  k_convert<<<64, 256, 0, stream>>>(Wg1, Wg1_bf, HDIM * HDIM);
  k_convert<<<4096, 256, 0, stream>>>(Wfc, Wfc_bf, NUM_ITEMS * HDIM);

  // Zero accumulation scratch every call (ws is not re-poisoned between replays)
  k_zero<<<8192, 256, 0, stream>>>(m, N_NODES * HDIM);
  k_zero<<<256, 256, 0, stream>>>(cnt, N_NODES);

  // Stage 1: embeddings + message linear
  k_message<<<N_NODES / 16, 256, 0, stream>>>(cat, sub, el, br, pid, price,
                                              ecat, esub, eel, ebr, eitem,
                                              Wm_bf, Wm, bm, h);
  // Stage 2: scatter-mean
  k_edge_count<<<N_EDGES / 256, 256, 0, stream>>>(ei, cnt);
  k_edge_scatter<<<(N_EDGES * HDIM) / 256, 256, 0, stream>>>(ei, h, m);
  k_normalize<<<(N_NODES * HDIM) / 256, 256, 0, stream>>>(m, cnt);

  // Stage 3: GRU (in-place h update)
  k_gru<<<N_NODES / 16, 256, 0, stream>>>(m, h, Wih_bf, Whh_bf, b_ih, b_hh);

  // Stage 4: attention gate
  k_attn<<<N_NODES / 16, 256, 0, stream>>>(h, Wg1_bf, bg1, Wg2, bg2, gate);

  // Stage 5: segment softmax + pooling
  k_softpool<<<N_GRAPHS, 128, 0, stream>>>(gate, h, pooled);

  // Stage 6: final FC -> (4096, 50000)
  dim3 fcgrid(N_GRAPHS / 16, (NUM_ITEMS + 127) / 128);
  k_fc<<<fcgrid, 256, 0, stream>>>(pooled, Wfc_bf, bfc, out);
}
